// DFlashAttentionANECache_46497315946512
// MI455X (gfx1250) — compile-verified, matrix operand order checked by reading, not run on hardware
//
#include <hip/hip_runtime.h>
#include <hip/hip_bf16.h>
#include <math.h>

// ---------------- problem constants ----------------
#define HDIM   2048
#define LQ     16
#define SCTX   2048
#define TKV    2064
#define DHEAD  128
#define NH     32
#define NKV    8
#define REP    4
#define SLEN   32768
#define EPSLN  1e-6f
#define SCALE  0.08838834764831845f  // 128^-0.5
#define NSPLIT 64
#define CHUNK  (SLEN / NSPLIT)       // 512
#define NITER  (CHUNK / 32)          // 16

typedef __attribute__((ext_vector_type(16))) _Float16 v16h;
typedef __attribute__((ext_vector_type(8)))  float    v8f;

// ---------------- WMMA fragment helpers (per CDNA5 ISA 16-bit layouts) ----
// A/B 16x32 f16 fragment: lane&15 selects row (A: M, B: N); within a lane the
// 16 halfs map to K = (lane>=16 ? 8 : 0) + (h&7) + (h>=8 ? 16 : 0).
__device__ __forceinline__ int frag_col(int h, int lane) {
  return ((lane >> 4) << 3) + (h & 7) + ((h >> 3) << 4);
}

__device__ __forceinline__ v16h pack16(float4 a0, float4 a1, float4 b0, float4 b1) {
  v16h f;
  f[0] = (_Float16)a0.x; f[1] = (_Float16)a0.y; f[2] = (_Float16)a0.z; f[3] = (_Float16)a0.w;
  f[4] = (_Float16)a1.x; f[5] = (_Float16)a1.y; f[6] = (_Float16)a1.z; f[7] = (_Float16)a1.w;
  f[8] = (_Float16)b0.x; f[9] = (_Float16)b0.y; f[10] = (_Float16)b0.z; f[11] = (_Float16)b0.w;
  f[12] = (_Float16)b1.x; f[13] = (_Float16)b1.y; f[14] = (_Float16)b1.z; f[15] = (_Float16)b1.w;
  return f;
}

// vectorized row-major fragment: element (k, lane&15) = base[(r0+lane&15)*ld + c0 + k]
// two contiguous 8-float runs -> 4x b128 loads
__device__ __forceinline__ v16h load_frag16v(const float* base, int ld, int r0, int c0, int lane) {
  const float* rp = base + (size_t)(r0 + (lane & 15)) * ld + c0 + ((lane >> 4) << 3);
  return pack16(*(const float4*)rp, *(const float4*)(rp + 4),
                *(const float4*)(rp + 16), *(const float4*)(rp + 20));
}

// same pattern but from an LDS tile (ds_load_b128-width reads)
__device__ __forceinline__ v16h frag_lds_rows(const float* lds, int ld, int r0, int c0, int lane) {
  const float* rp = lds + (r0 + (lane & 15)) * ld + c0 + ((lane >> 4) << 3);
  return pack16(*(const float4*)rp, *(const float4*)(rp + 4),
                *(const float4*)(rp + 16), *(const float4*)(rp + 20));
}

// transposed fragment from LDS: element (k, n) = lds[k*ld + c0 + lane&15]
__device__ __forceinline__ v16h frag_lds_T(const float* lds, int ld, int c0, int lane) {
  const float* cp = lds + c0 + (lane & 15);
  v16h f;
#pragma unroll
  for (int h = 0; h < 16; ++h) f[h] = (_Float16)cp[frag_col(h, lane) * ld];
  return f;
}

__device__ __forceinline__ v8f wmma_f16(v16h a, v16h b, v8f c) {
  return __builtin_amdgcn_wmma_f32_16x16x32_f16(false, a, false, b, (short)0, c, false, false);
}

__device__ __forceinline__ float wave_sum32(float v) {
#pragma unroll
  for (int off = 1; off < 32; off <<= 1) v += __shfl_xor(v, off, 32);
  return v;
}

// async copy of one 32x128 f32 chunk (16KB) into LDS: 8 instructions/wave,
// 128 threads x 8 x 16B. Tracked on ASYNCcnt (8 per wave per chunk).
__device__ __forceinline__ void async_chunk(const float* gbase, int kb0, unsigned ldsoff, int tid) {
#pragma unroll
  for (int j = 0; j < 8; ++j) {
    int i = j * 128 + tid;
    int row = i >> 5;
    int col = (i & 31) << 2;  // float index, 16B granules
    unsigned loff = ldsoff + (unsigned)((row * DHEAD + col) * 4);
    unsigned long long g =
        (unsigned long long)(uintptr_t)(gbase + (size_t)(kb0 + row) * DHEAD + col);
    asm volatile("global_load_async_to_lds_b128 %0, %1, off" :: "v"(loff), "v"(g) : "memory");
  }
}

// ---------------- 1) cache pass-through copy ----------------
__global__ void copy_cache_kernel(const float4* __restrict__ src, float4* __restrict__ dst, size_t n4) {
  size_t i = (size_t)blockIdx.x * blockDim.x + threadIdx.x;
  size_t stride = (size_t)gridDim.x * blockDim.x;
  for (; i < n4; i += stride) dst[i] = src[i];
}

// ---------------- 2) generic WMMA GEMM: C[M x N] = A[M x K] * W[N x K]^T ----
__global__ __launch_bounds__(128) void gemm_nt_wmma(const float* __restrict__ A0, int rows0,
                                                    const float* __restrict__ A1, int K,
                                                    const float* __restrict__ W,
                                                    float* __restrict__ C, int N) {
  const int lane = threadIdx.x & 31;
  const int wv = threadIdx.x >> 5;
  const int n0 = (blockIdx.x * 4 + wv) * 16;
  const int m0 = blockIdx.y * 16;
  const int row = m0 + (lane & 15);
  const float* rp = ((row < rows0) ? (A0 + (size_t)row * K)
                                   : (A1 + (size_t)(row - rows0) * K)) + ((lane >> 4) << 3);
  const float* wp = W + (size_t)(n0 + (lane & 15)) * K + ((lane >> 4) << 3);
  v8f acc = {};
  for (int k0 = 0; k0 < K; k0 += 32) {
    v16h a = pack16(*(const float4*)(rp + k0), *(const float4*)(rp + k0 + 4),
                    *(const float4*)(rp + k0 + 16), *(const float4*)(rp + k0 + 20));
    v16h b = pack16(*(const float4*)(wp + k0), *(const float4*)(wp + k0 + 4),
                    *(const float4*)(wp + k0 + 16), *(const float4*)(wp + k0 + 20));
    acc = wmma_f16(a, b, acc);
  }
  const int n = lane & 15;
  const int hs = (lane >> 4) << 3;
#pragma unroll
  for (int r = 0; r < 8; ++r)
    C[(size_t)(m0 + r + hs) * N + n0 + n] = acc[r];
}

// ---------------- 3) LN + RoPE on Q (in place, q layout [L][NH*D]) --------
__global__ __launch_bounds__(128) void ln_rope_q_kernel(float* __restrict__ q,
                                                        const float* __restrict__ w,
                                                        const float* __restrict__ cosq,
                                                        const float* __restrict__ sinq) {
  const int lane = threadIdx.x & 31;
  const int row = blockIdx.x * 4 + (threadIdx.x >> 5);  // [0, L*NH)
  const int tok = row / NH;
  float* rp = q + (size_t)row * DHEAD;
  float v[4];
#pragma unroll
  for (int j = 0; j < 4; ++j) v[j] = rp[lane + 32 * j];
  float mean = wave_sum32(v[0] + v[1] + v[2] + v[3]) * (1.0f / DHEAD);
  float xc[4], var = 0.f;
#pragma unroll
  for (int j = 0; j < 4; ++j) { xc[j] = v[j] - mean; var += xc[j] * xc[j]; }
  var = wave_sum32(var) * (1.0f / DHEAD);
  const float rs = rsqrtf(var + EPSLN);
  float xn[4];
#pragma unroll
  for (int j = 0; j < 4; ++j) xn[j] = xc[j] * rs * w[lane + 32 * j];
#pragma unroll
  for (int j = 0; j < 4; ++j) {
    int d = lane + 32 * j;
    float rot = (j < 2) ? -xn[j + 2] : xn[j - 2];
    rp[d] = xn[j] * cosq[(size_t)tok * DHEAD + d] + rot * sinq[(size_t)tok * DHEAD + d];
  }
}

// ---------------- 4) LN + RoPE on K, scatter K/V into out-cache -----------
__global__ __launch_bounds__(128) void ln_rope_kv_kernel(const float* __restrict__ kraw,
                                                         const float* __restrict__ vraw,
                                                         const float* __restrict__ w,
                                                         const float* __restrict__ cosk,
                                                         const float* __restrict__ sink,
                                                         float* __restrict__ cache,
                                                         const int* __restrict__ posp) {
  const int lane = threadIdx.x & 31;
  const int row = blockIdx.x * 4 + (threadIdx.x >> 5);  // [0, T*NKV)
  const int t = row / NKV;
  const int kvh = row % NKV;
  const int pos = *posp;
  const float* rp = kraw + (size_t)row * DHEAD;
  float v[4];
#pragma unroll
  for (int j = 0; j < 4; ++j) v[j] = rp[lane + 32 * j];
  float mean = wave_sum32(v[0] + v[1] + v[2] + v[3]) * (1.0f / DHEAD);
  float xc[4], var = 0.f;
#pragma unroll
  for (int j = 0; j < 4; ++j) { xc[j] = v[j] - mean; var += xc[j] * xc[j]; }
  var = wave_sum32(var) * (1.0f / DHEAD);
  const float rs = rsqrtf(var + EPSLN);
  float xn[4];
#pragma unroll
  for (int j = 0; j < 4; ++j) xn[j] = xc[j] * rs * w[lane + 32 * j];
  const size_t kout = ((size_t)kvh * SLEN + pos + t) * DHEAD;
  const size_t vout = ((size_t)(NKV + kvh) * SLEN + pos + t) * DHEAD;
  const float* vp = vraw + (size_t)row * DHEAD;
#pragma unroll
  for (int j = 0; j < 4; ++j) {
    int d = lane + 32 * j;
    float rot = (j < 2) ? -xn[j + 2] : xn[j - 2];
    cache[kout + d] = xn[j] * cosk[(size_t)t * DHEAD + d] + rot * sink[(size_t)t * DHEAD + d];
    cache[vout + d] = vp[d];
  }
}

// ---------------- 5) split-K flash attention partials ---------------------
// grid = (NSPLIT, NKV); 4 waves, wave = rep head. K double-buffered and V
// single-buffered in LDS via async-to-LDS; each KV element read from HBM once
// per workgroup (not once per wave).
__global__ __launch_bounds__(128) void attn_partial_kernel(const float* __restrict__ qws,
                                                           const float* __restrict__ cache,
                                                           const float* __restrict__ mask,
                                                           float* __restrict__ pout,
                                                           float* __restrict__ pm,
                                                           float* __restrict__ pl) {
  const int tid = threadIdx.x;
  const int lane = tid & 31;
  const int wave = tid >> 5;  // rep index
  const int kvh = blockIdx.y;
  const int split = blockIdx.x;
  const int h = kvh * REP + wave;
  const float* kbase = cache + (size_t)kvh * SLEN * DHEAD;
  const float* vbase = cache + (size_t)(NKV + kvh) * SLEN * DHEAD;
  const float* qb = qws + (size_t)h * DHEAD;  // row stride NH*D

  __shared__ float kch[2][32][DHEAD];   // 32 KB
  __shared__ float vch[32][DHEAD];      // 16 KB
  __shared__ _Float16 plds[4][16][32];  // 4 KB
  const unsigned koff0 = (unsigned)(uintptr_t)&kch[0][0][0];
  const unsigned koff1 = (unsigned)(uintptr_t)&kch[1][0][0];
  const unsigned voff  = (unsigned)(uintptr_t)&vch[0][0];

  v16h qa[4];
#pragma unroll
  for (int t = 0; t < 4; ++t) qa[t] = load_frag16v(qb, NH * DHEAD, 0, t * 32, lane);

  v8f acc[8];
  float mrow[8], lrow[8];
#pragma unroll
  for (int f = 0; f < 8; ++f) acc[f] = (v8f){};
#pragma unroll
  for (int r = 0; r < 8; ++r) { mrow[r] = -1e30f; lrow[r] = 0.f; }

  const int n = lane & 15;
  const int hs = (lane >> 4) << 3;
  const int kstart = split * CHUNK;

  // prologue: K chunk 0 in flight
  async_chunk(kbase, kstart, koff0, tid);

  for (int it = 0; it < NITER; ++it) {
    const int kb0 = kstart + it * 32;
    const float* kc = &kch[it & 1][0][0];
    // V(it) now; K(it+1) prefetch. Per-wave ASYNCcnt groups of 8, in order.
    async_chunk(vbase, kb0, voff, tid);
    if (it + 1 < NITER) {
      async_chunk(kbase, kb0 + 32, (it & 1) ? koff0 : koff1, tid);
      asm volatile("s_wait_asynccnt 0x10" ::: "memory");  // K(it) landed
    } else {
      asm volatile("s_wait_asynccnt 0x8" ::: "memory");
    }
    __syncthreads();

    // ---- S = Q * K^T over D=128, two 16x16 column tiles ----
    v8f s0 = (v8f){}, s1 = (v8f){};
#pragma unroll
    for (int t = 0; t < 4; ++t) {
      v16h b0 = frag_lds_rows(kc, DHEAD, 0, t * 32, lane);
      v16h b1 = frag_lds_rows(kc, DHEAD, 16, t * 32, lane);
      s0 = wmma_f16(qa[t], b0, s0);
      s1 = wmma_f16(qa[t], b1, s1);
    }
    // ---- online softmax per row (row = r + hs, 16 lanes per half) ----
    float p0[8], p1[8];
#pragma unroll
    for (int r = 0; r < 8; ++r) {
      int m = r + hs;  // token row
      float v0 = s0[r] * SCALE + mask[(size_t)m * SLEN + kb0 + n];
      float v1 = s1[r] * SCALE + mask[(size_t)m * SLEN + kb0 + 16 + n];
      float mx = fmaxf(v0, v1);
#pragma unroll
      for (int off = 1; off < 16; off <<= 1) mx = fmaxf(mx, __shfl_xor(mx, off, 32));
      float mnew = fmaxf(mrow[r], mx);
      float fac = __expf(mrow[r] - mnew);
      mrow[r] = mnew;
      p0[r] = __expf(v0 - mnew);
      p1[r] = __expf(v1 - mnew);
      float ps = p0[r] + p1[r];
#pragma unroll
      for (int off = 1; off < 16; off <<= 1) ps += __shfl_xor(ps, off, 32);
      lrow[r] = lrow[r] * fac + ps;
#pragma unroll
      for (int f = 0; f < 8; ++f) acc[f][r] *= fac;
    }
    // ---- relayout P (C-layout -> A-frag layout) through per-wave LDS ----
#pragma unroll
    for (int r = 0; r < 8; ++r) {
      plds[wave][r + hs][n] = (_Float16)p0[r];
      plds[wave][r + hs][16 + n] = (_Float16)p1[r];
    }
    asm volatile("s_wait_dscnt 0x0" ::: "memory");
    v16h pa;
#pragma unroll
    for (int hh = 0; hh < 16; ++hh) pa[hh] = plds[wave][lane & 15][frag_col(hh, lane)];

    // V(it) must have landed before PV
    if (it + 1 < NITER) {
      asm volatile("s_wait_asynccnt 0x8" ::: "memory");
    } else {
      asm volatile("s_wait_asynccnt 0x0" ::: "memory");
    }
    __syncthreads();
    // ---- O += P * V  (8 n-tiles over D=128) ----
#pragma unroll
    for (int f = 0; f < 8; ++f) {
      v16h vb = frag_lds_T(&vch[0][0], DHEAD, f * 16, lane);
      acc[f] = wmma_f16(pa, vb, acc[f]);
    }
    __syncthreads();
  }

  const size_t pidx = (size_t)h * NSPLIT + split;
#pragma unroll
  for (int r = 0; r < 8; ++r) {
    int m = r + hs;
    if (n == 0) {
      pm[pidx * 16 + m] = mrow[r];
      pl[pidx * 16 + m] = lrow[r];
    }
#pragma unroll
    for (int f = 0; f < 8; ++f)
      pout[(pidx * 16 + m) * DHEAD + f * 16 + n] = acc[f][r];
  }
}

// ---------------- 6) LSE combine over splits ------------------------------
__global__ __launch_bounds__(128) void attn_reduce_kernel(const float* __restrict__ pout,
                                                          const float* __restrict__ pm,
                                                          const float* __restrict__ pl,
                                                          float* __restrict__ aout) {
  const int h = blockIdx.y;
  const int tok = blockIdx.x;
  const int d = threadIdx.x;  // 128
  float M = -1e30f;
  for (int s = 0; s < NSPLIT; ++s)
    M = fmaxf(M, pm[((size_t)h * NSPLIT + s) * 16 + tok]);
  float acc = 0.f, lsum = 0.f;
  for (int s = 0; s < NSPLIT; ++s) {
    size_t pidx = (size_t)h * NSPLIT + s;
    float wgt = __expf(pm[pidx * 16 + tok] - M);
    lsum += wgt * pl[pidx * 16 + tok];
    acc += wgt * pout[(pidx * 16 + tok) * DHEAD + d];
  }
  aout[(size_t)tok * (NH * DHEAD) + (size_t)h * DHEAD + d] = acc / lsum;
}

// ---------------- launch ---------------------------------------------------
extern "C" void kernel_launch(void* const* d_in, const int* in_sizes, int n_in,
                              void* d_out, int out_size, void* d_ws, size_t ws_size,
                              hipStream_t stream) {
  const float* x       = (const float*)d_in[0];
  const float* x_ctx   = (const float*)d_in[1];
  const float* cos_q   = (const float*)d_in[2];
  const float* sin_q   = (const float*)d_in[3];
  const float* cos_k   = (const float*)d_in[4];
  const float* sin_k   = (const float*)d_in[5];
  const float* kvcache = (const float*)d_in[6];
  const float* mask    = (const float*)d_in[7];
  const float* wq      = (const float*)d_in[8];
  const float* wk      = (const float*)d_in[9];
  const float* wv      = (const float*)d_in[10];
  const float* wo      = (const float*)d_in[11];
  const float* qn_w    = (const float*)d_in[12];
  const float* kn_w    = (const float*)d_in[13];
  const int*   posp    = (const int*)d_in[14];

  float* out_x     = (float*)d_out;                       // (L, H)
  float* out_cache = (float*)d_out + (size_t)LQ * HDIM;   // (2, NKV, SL, D)

  float* ws = (float*)d_ws;
  float* q_ws  = ws;                                // L  * NH*D
  float* k_raw = q_ws + (size_t)LQ * NH * DHEAD;    // T  * NKV*D
  float* v_raw = k_raw + (size_t)TKV * NKV * DHEAD;
  float* pout  = v_raw + (size_t)TKV * NKV * DHEAD; // NH*NSPLIT*16*D
  float* pm    = pout + (size_t)NH * NSPLIT * 16 * DHEAD;
  float* pl    = pm + (size_t)NH * NSPLIT * 16;
  float* aout  = pl + (size_t)NH * NSPLIT * 16;     // L * NH*D

  // 1) pass-through copy of the KV cache (bandwidth-dominant)
  size_t n4 = (size_t)2 * NKV * SLEN * DHEAD / 4;
  copy_cache_kernel<<<4096, 256, 0, stream>>>((const float4*)kvcache, (float4*)out_cache, n4);

  // 2) projections: Q = x@wq^T, Kraw/Vraw = concat(x_ctx,x)@w^T
  gemm_nt_wmma<<<dim3(NH * DHEAD / 64, LQ / 16), 128, 0, stream>>>(x, LQ, x, HDIM, wq, q_ws, NH * DHEAD);
  gemm_nt_wmma<<<dim3(NKV * DHEAD / 64, TKV / 16), 128, 0, stream>>>(x_ctx, SCTX, x, HDIM, wk, k_raw, NKV * DHEAD);
  gemm_nt_wmma<<<dim3(NKV * DHEAD / 64, TKV / 16), 128, 0, stream>>>(x_ctx, SCTX, x, HDIM, wv, v_raw, NKV * DHEAD);

  // 3) LN + RoPE
  ln_rope_q_kernel<<<LQ * NH / 4, 128, 0, stream>>>(q_ws, qn_w, cos_q, sin_q);
  ln_rope_kv_kernel<<<TKV * NKV / 4, 128, 0, stream>>>(k_raw, v_raw, kn_w, cos_k, sin_k, out_cache, posp);

  // 4) split-K flash attention over the full (updated) cache
  attn_partial_kernel<<<dim3(NSPLIT, NKV), 128, 0, stream>>>(q_ws, out_cache, mask, pout, pm, pl);
  attn_reduce_kernel<<<dim3(LQ, NH), 128, 0, stream>>>(pout, pm, pl, aout);

  // 5) output projection: out = aout(16x4096) @ wo(2048x4096)^T
  gemm_nt_wmma<<<dim3(HDIM / 64, LQ / 16), 128, 0, stream>>>(aout, LQ, aout, NH * DHEAD, wo, out_x, HDIM);
}
